// BiTreeLSTM_65171833749675
// MI455X (gfx1250) — matrix operand order repaired and checked by reading, not compiled, not run on hardware
//
#include <hip/hip_runtime.h>
#include <hip/hip_bf16.h>

typedef __attribute__((ext_vector_type(16))) _Float16 v16h;
typedef __attribute__((ext_vector_type(8)))  float    v8f;

#define LSEQ 256
#define BATCH 128
#define DDIM 256
#define HDIM 256
// concatenated projection width: [i|o|u] (3H) + [f] (H) = 1024
#define NW 1024

// ---------------------------------------------------------------------------
// A-operand gather for V_WMMA_F32_16X16X32_F16.
// 16-bit A 16x32 layout (ISA 7.12.2): lane m (&15) holds row M=m.
//   lanes 0-15 : K = {0..7, 16..23};  lanes 16-31: K = {8..15, 24..31}
//   half-pair j<4 -> K0 = 2j ; j>=4 -> K0 = 2j+8  (then +base8 for hi lanes)
// Source is f32; convert to f16 on the fly (v_cvt_pk_f16_f32).
// ---------------------------------------------------------------------------
__device__ __forceinline__ v16h load_a_f32(const float* row, int s, int base8) {
    v16h a;
#pragma unroll
    for (int j = 0; j < 8; ++j) {
        int k0 = s * 32 + base8 + ((j < 4) ? (2 * j) : (2 * j + 8));
        a[2 * j]     = (_Float16)row[k0];
        a[2 * j + 1] = (_Float16)row[k0 + 1];
    }
    return a;
}

__device__ __forceinline__ v16h load_a_guard(const float* row, int s, int base8, bool valid) {
    v16h a;
#pragma unroll
    for (int j = 0; j < 8; ++j) {
        int k0 = s * 32 + base8 + ((j < 4) ? (2 * j) : (2 * j + 8));
        a[2 * j]     = (_Float16)(valid ? row[k0]     : 0.0f);
        a[2 * j + 1] = (_Float16)(valid ? row[k0 + 1] : 0.0f);
    }
    return a;
}

__device__ __forceinline__ float sigmoidf_(float x) { return 1.0f / (1.0f + __expf(-x)); }

// ---------------------------------------------------------------------------
// Pack the four concatenated weight matrices [NW=1024 rows (N), 256 cols (K)]
// into the per-lane WMMA B layout, f32 -> f16:
//   packed[mat][ntile][kstep][lane][16 halves]
//   lane n (&15) holds column N = ntile*16+n ; lanes 0-15 K={0..7,16..23}, lanes>=16 K={8..15,24..31}.
// mat: 0 = Wx_dt (ioux|fx), 1 = Wh_dt (iouh|fh), 2 = Wx_td, 3 = Wh_td
// ---------------------------------------------------------------------------
__global__ void __launch_bounds__(256)
pack_weights_kernel(const float* dt_ioux_w, const float* dt_fx_w,
                    const float* dt_iouh_w, const float* dt_fh_w,
                    const float* td_ioux_w, const float* td_fx_w,
                    const float* td_iouh_w, const float* td_fh_w,
                    _Float16* packed) {
    int e = blockIdx.x * 256 + threadIdx.x;      // 0 .. 4*262144-1
    int mat   = e >> 18;
    int r     = e & 262143;
    int ntile = r >> 12;
    int s     = (r >> 9) & 7;
    int lane  = (r >> 4) & 31;
    int t     = r & 15;
    int n = ntile * 16 + (lane & 15);
    int k = s * 32 + ((lane >= 16) ? 8 : 0) + ((t < 8) ? t : (t + 8));
    const float* lo; const float* hi;
    switch (mat) {
        case 0:  lo = dt_ioux_w; hi = dt_fx_w; break;
        case 1:  lo = dt_iouh_w; hi = dt_fh_w; break;
        case 2:  lo = td_ioux_w; hi = td_fx_w; break;
        default: lo = td_iouh_w; hi = td_fh_w; break;
    }
    float v = (n < 768) ? lo[(size_t)n * 256 + k] : hi[(size_t)(n - 768) * 256 + k];
    packed[e] = (_Float16)v;
}

// bsum[dir][1024] = (ioux_b+iouh_b | fx_b+fh_b)  (folded once into XW)
__global__ void __launch_bounds__(256)
pack_bias_kernel(const float* dt_ioux_b, const float* dt_iouh_b,
                 const float* dt_fx_b,   const float* dt_fh_b,
                 const float* td_ioux_b, const float* td_iouh_b,
                 const float* td_fx_b,   const float* td_fh_b,
                 float* bsum) {
    int e = blockIdx.x * 256 + threadIdx.x;      // 0..2047
    int dir = e >> 10;
    int n   = e & 1023;
    const float* ia = dir ? td_ioux_b : dt_ioux_b;
    const float* ih = dir ? td_iouh_b : dt_iouh_b;
    const float* fa = dir ? td_fx_b   : dt_fx_b;
    const float* fh = dir ? td_fh_b   : dt_fh_b;
    bsum[e] = (n < 768) ? (ia[n] + ih[n]) : (fa[n - 768] + fh[n - 768]);
}

// ---------------------------------------------------------------------------
// Phase 1: XW[dir][32768, 1024] (f16) = inputs[32768, 256] @ Wx[dir].T + bsum
// All 8 waves of a block share one 16-row M-tile: wave w gathers/converts
// k-step w of the A tile into LDS (per-lane WMMA layout), then every wave
// consumes it via ds_load. 32 WMMAs / wave.
// ---------------------------------------------------------------------------
__global__ void __launch_bounds__(256)
xproj_kernel(const float* inputs, const _Float16* pkWx_dt, const _Float16* pkWx_td,
             const float* bsum, _Float16* xw_dt, _Float16* xw_td) {
    __shared__ v16h lds_a[8][32];                            // [kstep][lane], 8 KB
    const int lane   = threadIdx.x & 31;
    const int wave   = threadIdx.x >> 5;
    const int gw     = blockIdx.x * 8 + wave;                // 0..32767
    const int mtile  = gw >> 4;                              // 0..2047 (same for whole block)
    const int ngroup = gw & 15;
    const int dir    = blockIdx.y;
    const _Float16* pk = dir ? pkWx_td : pkWx_dt;
    _Float16* xw       = dir ? xw_td   : xw_dt;
    const float* bias  = bsum + dir * 1024;
    const int base8 = (lane >= 16) ? 8 : 0;
    const float* arow = inputs + (size_t)(mtile * 16 + (lane & 15)) * DDIM;
    const v16h* bp = (const v16h*)pk;

    // cooperative A stage: wave w handles k-step w
    lds_a[wave][lane] = load_a_f32(arow, wave, base8);
    __syncthreads();

    v8f acc0 = {}, acc1 = {}, acc2 = {}, acc3 = {};
#pragma unroll
    for (int s = 0; s < 8; ++s) {
        v16h a  = lds_a[s][lane];
        v16h b0 = bp[((ngroup * 4 + 0) * 8 + s) * 32 + lane];
        v16h b1 = bp[((ngroup * 4 + 1) * 8 + s) * 32 + lane];
        v16h b2 = bp[((ngroup * 4 + 2) * 8 + s) * 32 + lane];
        v16h b3 = bp[((ngroup * 4 + 3) * 8 + s) * 32 + lane];
        acc0 = __builtin_amdgcn_wmma_f32_16x16x32_f16(false, a, false, b0, (short)0, acc0, false, false);
        acc1 = __builtin_amdgcn_wmma_f32_16x16x32_f16(false, a, false, b1, (short)0, acc1, false, false);
        acc2 = __builtin_amdgcn_wmma_f32_16x16x32_f16(false, a, false, b2, (short)0, acc2, false, false);
        acc3 = __builtin_amdgcn_wmma_f32_16x16x32_f16(false, a, false, b3, (short)0, acc3, false, false);
    }
    // C/D layout: VGPR v, lanes<16 -> M=v ; lanes>=16 -> M=v+8 ; N=lane&15
    const int mrow_off = (lane < 16) ? 0 : 8;
    v8f accs[4] = {acc0, acc1, acc2, acc3};
#pragma unroll
    for (int q = 0; q < 4; ++q) {
        int col = (ngroup * 4 + q) * 16 + (lane & 15);
        float bv = bias[col];
#pragma unroll
        for (int v = 0; v < 8; ++v) {
            size_t row = (size_t)(mtile * 16 + v + mrow_off);
            xw[row * NW + col] = (_Float16)(accs[q][v] + bv);
        }
    }
}

// ---------------------------------------------------------------------------
// Phase 2a (bottom-up, one launch per scan step s_step):
//   sum_h = acc_h[b, idx] ;  pre = sum_h @ Wh.T + XW[row(idx,b)]
//   gates -> c,h ; write state to d_out ; scatter-add (c,h) into acc at parent.
// Wave (mtile 0..7, j 0..15) computes the matching i/o/u/f 16x16 tiles for
// H-columns j*16..j*16+15, so the epilogue is fully local to the wave.
// A tile (shared by the block's 8 waves) staged once through LDS.
// ---------------------------------------------------------------------------
__global__ void __launch_bounds__(256)
dt_step_kernel(const _Float16* xw, const _Float16* pkWh,
               const int* indexes, const int* parents,
               float* acc_c, float* acc_h,
               float* cells, float* hiddens, int s_step) {
    __shared__ v16h lds_a[8][32];
    const int lane  = threadIdx.x & 31;
    const int wave  = threadIdx.x >> 5;
    const int gw    = blockIdx.x * 8 + wave;                // 0..127
    const int mtile = gw >> 4;                              // 0..7 (same per block)
    const int j     = gw & 15;
    const int base8 = (lane >= 16) ? 8 : 0;

    const int bA    = mtile * 16 + (lane & 15);
    const int nodeA = indexes[s_step * BATCH + bA];
    const float* arow = acc_h + ((size_t)bA * (LSEQ + 1) + nodeA) * HDIM;
    __builtin_prefetch(xw + ((size_t)nodeA * BATCH + bA) * NW, 0, 0);
    const v16h* bp = (const v16h*)pkWh;

    lds_a[wave][lane] = load_a_f32(arow, wave, base8);
    __syncthreads();

    v8f ai = {}, ao = {}, au = {}, af = {};
#pragma unroll
    for (int s = 0; s < 8; ++s) {
        v16h a  = lds_a[s][lane];
        v16h bi = bp[((j +  0) * 8 + s) * 32 + lane];
        v16h bo = bp[((j + 16) * 8 + s) * 32 + lane];
        v16h bu = bp[((j + 32) * 8 + s) * 32 + lane];
        v16h bf = bp[((j + 48) * 8 + s) * 32 + lane];
        ai = __builtin_amdgcn_wmma_f32_16x16x32_f16(false, a, false, bi, (short)0, ai, false, false);
        ao = __builtin_amdgcn_wmma_f32_16x16x32_f16(false, a, false, bo, (short)0, ao, false, false);
        au = __builtin_amdgcn_wmma_f32_16x16x32_f16(false, a, false, bu, (short)0, au, false, false);
        af = __builtin_amdgcn_wmma_f32_16x16x32_f16(false, a, false, bf, (short)0, af, false, false);
    }

    const int mrow_off = (lane < 16) ? 0 : 8;
    const int hcol = j * 16 + (lane & 15);
#pragma unroll
    for (int v = 0; v < 8; ++v) {
        int b    = mtile * 16 + v + mrow_off;
        int node = indexes[s_step * BATCH + b];
        int par  = parents[node * BATCH + b];      // root -> LSEQ (dummy slot)
        const _Float16* xr = xw + ((size_t)node * BATCH + b) * NW;
        float pi = ai[v] + (float)xr[hcol];
        float po = ao[v] + (float)xr[256 + hcol];
        float pu = au[v] + (float)xr[512 + hcol];
        float pf = af[v] + (float)xr[768 + hcol];
        float sc = acc_c[((size_t)b * (LSEQ + 1) + node) * HDIM + hcol];
        float ig = sigmoidf_(pi), og = sigmoidf_(po);
        float ug = tanhf(pu),     fg = sigmoidf_(pf);
        float c = ig * ug + fg * sc;
        float h = og * tanhf(c);
        size_t ocell = ((size_t)node * BATCH + b) * (2 * HDIM) + hcol; // dt cols 0..255
        cells[ocell]   = c;
        hiddens[ocell] = h;
        size_t pacc = ((size_t)b * (LSEQ + 1) + par) * HDIM + hcol;
        acc_c[pacc] += c;
        acc_h[pacc] += h;
    }
}

// ---------------------------------------------------------------------------
// Phase 2b (top-down): parent (c,h) gathered straight from d_out (parent is
// always written before child); root's parent slot LSEQ reads guarded zeros.
// ---------------------------------------------------------------------------
__global__ void __launch_bounds__(256)
td_step_kernel(const _Float16* xw, const _Float16* pkWh,
               const int* indexes, const int* parents,
               float* cells, float* hiddens, int s_step) {
    __shared__ v16h lds_a[8][32];
    const int lane  = threadIdx.x & 31;
    const int wave  = threadIdx.x >> 5;
    const int gw    = blockIdx.x * 8 + wave;
    const int mtile = gw >> 4;
    const int j     = gw & 15;
    const int base8 = (lane >= 16) ? 8 : 0;

    const int bA    = mtile * 16 + (lane & 15);
    const int nodeA = indexes[s_step * BATCH + bA];
    const int parA  = parents[nodeA * BATCH + bA];
    const bool vA   = (parA < LSEQ);
    const float* arow = hiddens + ((size_t)(vA ? parA : 0) * BATCH + bA) * (2 * HDIM) + HDIM;
    const v16h* bp = (const v16h*)pkWh;

    lds_a[wave][lane] = load_a_guard(arow, wave, base8, vA);
    __syncthreads();

    v8f ai = {}, ao = {}, au = {}, af = {};
#pragma unroll
    for (int s = 0; s < 8; ++s) {
        v16h a  = lds_a[s][lane];
        v16h bi = bp[((j +  0) * 8 + s) * 32 + lane];
        v16h bo = bp[((j + 16) * 8 + s) * 32 + lane];
        v16h bu = bp[((j + 32) * 8 + s) * 32 + lane];
        v16h bf = bp[((j + 48) * 8 + s) * 32 + lane];
        ai = __builtin_amdgcn_wmma_f32_16x16x32_f16(false, a, false, bi, (short)0, ai, false, false);
        ao = __builtin_amdgcn_wmma_f32_16x16x32_f16(false, a, false, bo, (short)0, ao, false, false);
        au = __builtin_amdgcn_wmma_f32_16x16x32_f16(false, a, false, bu, (short)0, au, false, false);
        af = __builtin_amdgcn_wmma_f32_16x16x32_f16(false, a, false, bf, (short)0, af, false, false);
    }

    const int mrow_off = (lane < 16) ? 0 : 8;
    const int hcol = j * 16 + (lane & 15);
#pragma unroll
    for (int v = 0; v < 8; ++v) {
        int b    = mtile * 16 + v + mrow_off;
        int node = indexes[s_step * BATCH + b];
        int par  = parents[node * BATCH + b];
        bool valid = (par < LSEQ);
        const _Float16* xr = xw + ((size_t)node * BATCH + b) * NW;
        float pi = ai[v] + (float)xr[hcol];
        float po = ao[v] + (float)xr[256 + hcol];
        float pu = au[v] + (float)xr[512 + hcol];
        float pf = af[v] + (float)xr[768 + hcol];
        float pc = valid ? cells[((size_t)par * BATCH + b) * (2 * HDIM) + HDIM + hcol] : 0.0f;
        float ig = sigmoidf_(pi), og = sigmoidf_(po);
        float ug = tanhf(pu),     fg = sigmoidf_(pf);
        float c = ig * ug + fg * pc;
        float h = og * tanhf(c);
        size_t ocell = ((size_t)node * BATCH + b) * (2 * HDIM) + HDIM + hcol; // td cols 256..511
        cells[ocell]   = c;
        hiddens[ocell] = h;
    }
}

// ---------------------------------------------------------------------------
// Workspace layout (bytes):
//   [0)            XW_dt   f16 32768x1024              = 64 MiB
//   [64Mi)         XW_td   f16                         = 64 MiB
//   [128Mi)        acc_c   f32 [B,L+1,H]               = 33,685,504
//   [+33.7M)       acc_h   f32 [B,L+1,H]               = 33,685,504
//   [+...)         packed weights f16, 4 x 262144      = 2 MiB
//   [+2Mi)         bsum    f32 2x1024                  = 8 KiB
// ---------------------------------------------------------------------------
extern "C" void kernel_launch(void* const* d_in, const int* in_sizes, int n_in,
                              void* d_out, int out_size, void* d_ws, size_t ws_size,
                              hipStream_t stream) {
    const float* inputs  = (const float*)d_in[0];
    const int*   indexes = (const int*)d_in[1];
    const int*   parents = (const int*)d_in[2];
    const float* dt_ioux_w = (const float*)d_in[3];
    const float* dt_ioux_b = (const float*)d_in[4];
    const float* dt_iouh_w = (const float*)d_in[5];
    const float* dt_iouh_b = (const float*)d_in[6];
    const float* dt_fx_w   = (const float*)d_in[7];
    const float* dt_fx_b   = (const float*)d_in[8];
    const float* dt_fh_w   = (const float*)d_in[9];
    const float* dt_fh_b   = (const float*)d_in[10];
    const float* td_ioux_w = (const float*)d_in[11];
    const float* td_ioux_b = (const float*)d_in[12];
    const float* td_iouh_w = (const float*)d_in[13];
    const float* td_iouh_b = (const float*)d_in[14];
    const float* td_fx_w   = (const float*)d_in[15];
    const float* td_fx_b   = (const float*)d_in[16];
    const float* td_fh_w   = (const float*)d_in[17];
    const float* td_fh_b   = (const float*)d_in[18];

    char* ws = (char*)d_ws;
    _Float16* xw_dt  = (_Float16*)(ws);
    _Float16* xw_td  = (_Float16*)(ws + 67108864);
    float*    acc_c  = (float*)   (ws + 134217728);
    float*    acc_h  = (float*)   (ws + 167903232);
    _Float16* packed = (_Float16*)(ws + 201588736);
    float*    bsum   = (float*)   (ws + 203685888);

    float* cells   = (float*)d_out;
    float* hiddens = cells + (size_t)LSEQ * BATCH * 2 * HDIM; // +16,777,216

    // zero child-sum accumulators (acc_c and acc_h are contiguous)
    hipMemsetAsync(acc_c, 0, (size_t)2 * 33685504, stream);

    pack_weights_kernel<<<4096, 256, 0, stream>>>(
        dt_ioux_w, dt_fx_w, dt_iouh_w, dt_fh_w,
        td_ioux_w, td_fx_w, td_iouh_w, td_fh_w, packed);
    pack_bias_kernel<<<8, 256, 0, stream>>>(
        dt_ioux_b, dt_iouh_b, dt_fx_b, dt_fh_b,
        td_ioux_b, td_iouh_b, td_fx_b, td_fh_b, bsum);

    // Phase 1: both directions' input projections in one grid (y = dir)
    xproj_kernel<<<dim3(4096, 2), 256, 0, stream>>>(
        inputs, packed /*Wx_dt*/, packed + 2 * 262144 /*Wx_td*/, bsum, xw_dt, xw_td);

    // Phase 2a: bottom-up scan (children before parents)
    for (int t = 0; t < LSEQ; ++t)
        dt_step_kernel<<<16, 256, 0, stream>>>(
            xw_dt, packed + 1 * 262144 /*Wh_dt*/, indexes, parents,
            acc_c, acc_h, cells, hiddens, t);

    // Phase 2b: top-down scan (reference scans xs in reverse: t = L-1 .. 0)
    for (int t = LSEQ - 1; t >= 0; --t)
        td_step_kernel<<<16, 256, 0, stream>>>(
            xw_td, packed + 3 * 262144 /*Wh_td*/, indexes, parents,
            cells, hiddens, t);
}
